// RegionProposalNetwork_824633721069
// MI455X (gfx1250) — matrix operand order repaired and checked by reading, not compiled, not run on hardware
//
#include <hip/hip_runtime.h>
#include <hip/hip_bf16.h>
#include <math.h>

typedef __attribute__((ext_vector_type(16))) _Float16 v16h;
typedef __attribute__((ext_vector_type(8)))  _Float16 v8h;
typedef __attribute__((ext_vector_type(8)))  float    v8f;
typedef int v4i __attribute__((vector_size(16)));   // matches builtin's V4i parameter type

union FragU { unsigned u[8]; v16h v; };

#define STRH 40   // LDS row stride in halfs: 80B rows -> 16B-aligned row starts, dword-aligned pairs

// ---- CDNA5 async global->LDS staging (guarded; falls back to wide register copy) ----
#if defined(__gfx1250__) && __has_builtin(__builtin_amdgcn_global_load_async_to_lds_b128)
#define HAS_ASYNC_LDS 1
#else
#define HAS_ASYNC_LDS 0
#endif

__device__ __forceinline__ void stage16B(_Float16* lds_dst, const _Float16* gsrc) {
#if HAS_ASYNC_LDS
    __builtin_amdgcn_global_load_async_to_lds_b128(
        (__attribute__((address_space(1))) v4i*)gsrc,
        (__attribute__((address_space(3))) v4i*)lds_dst, 0, 0);
#else
    *(v8h*)lds_dst = *(const v8h*)gsrc;
#endif
}
__device__ __forceinline__ void stage_wait() {
#if HAS_ASYNC_LDS
#if __has_builtin(__builtin_amdgcn_s_wait_asynccnt)
    __builtin_amdgcn_s_wait_asynccnt(0);
#else
    asm volatile("s_wait_asynccnt 0x0" ::: "memory");
#endif
#endif
}

// ---------------- conversion kernels ----------------
__global__ void f32_to_f16_kernel(const float* __restrict__ in, _Float16* __restrict__ out, int n) {
    int i = blockIdx.x * blockDim.x + threadIdx.x;
    if (i < n) out[i] = (_Float16)in[i];
}

// w1 (OIHW, 512x512x3x3 f32) -> wt[dyx][k][c] f16 (contiguous in c for wide staging)
__global__ void transform_w1_kernel(const float* __restrict__ w1, _Float16* __restrict__ wt) {
    int o = blockIdx.x * blockDim.x + threadIdx.x;          // 0 .. 512*512*9-1
    if (o >= 512 * 512 * 9) return;
    int c   = o & 511;
    int k   = (o >> 9) & 511;
    int dyx = o >> 18;
    wt[o] = (_Float16)w1[((size_t)k * 512 + c) * 9 + dyx];
}

// combined 1x1 weights: rows 0..17 = ws, 18..53 = wl, 54..63 = 0   (64 x 512, f16)
__global__ void build_wc_kernel(const float* __restrict__ wsq, const float* __restrict__ wlq,
                                _Float16* __restrict__ wc) {
    int i = blockIdx.x * blockDim.x + threadIdx.x;
    if (i >= 64 * 512) return;
    int m = i >> 9, c = i & 511;
    float v = 0.f;
    if (m < 18)       v = wsq[m * 512 + c];
    else if (m < 54)  v = wlq[(m - 18) * 512 + c];
    wc[i] = (_Float16)v;
}

// ---------------- anchors ----------------
__global__ void anchors_kernel(float* __restrict__ anch) {
    int i = blockIdx.x * blockDim.x + threadIdx.x;
    if (i >= 36864) return;
    int a = i % 9, pix = i / 9;
    int y = pix >> 6, x = pix & 63;
    const float R[3] = {0.5f, 1.0f, 2.0f};
    const float S[3] = {8.f, 16.f, 32.f};
    float r = R[a / 3], s = S[a % 3];
    float h = 16.f * s * sqrtf(r);
    float w = 16.f * s * sqrtf(1.0f / r);
    float cy = y * 16.f + 8.f, cx = x * 16.f + 8.f;
    anch[(size_t)i * 4 + 0] = cy - 0.5f * h;
    anch[(size_t)i * 4 + 1] = cx - 0.5f * w;
    anch[(size_t)i * 4 + 2] = cy + 0.5f * h;
    anch[(size_t)i * 4 + 3] = cx + 0.5f * w;
}

// ---------------- WMMA fragment K maps (per CDNA5 ISA 7.12.2) ----------------
__device__ __forceinline__ int kmapA(int v, int half) {   // 16-bit A 16x32
    return ((v & 3) * 2) + ((v >> 2) * 16) + half * 8;
}
__device__ __forceinline__ int kmapB(int v, int half) {   // 16-bit B 32x16 (dense)
    return ((v & 3) * 2) + ((v >> 2) * 16) + half * 8;
}

// ---------------- 3x3 conv (implicit GEMM, WMMA) ----------------
// block: 256 threads (8 waves). tile: M=64 out-channels x N=64 pixels (one row).
// grid: (512/64 M-tiles, 64 rows, 8 images)
__global__ __launch_bounds__(256) void conv3x3_wmma_kernel(
    const _Float16* __restrict__ xf, const _Float16* __restrict__ wt,
    const float* __restrict__ b1, _Float16* __restrict__ cf) {
    __shared__ __align__(16) _Float16 As[64 * STRH];
    __shared__ __align__(16) _Float16 Bs[64 * STRH];
    const int tid  = threadIdx.x;
    const int m0   = blockIdx.x * 64;
    const int y    = blockIdx.y;
    const int b    = blockIdx.z;
    const int lane = tid & 31;
    const int wv   = tid >> 5;
    const int half = lane >> 4;
    const int l16  = lane & 15;
    const int mt   = wv & 3;          // M tile 0..3
    const int ng   = wv >> 2;         // N tile group 0..1 (2 tiles each)

    v8f acc0 = {}; v8f acc1 = {};

    for (int dyx = 0; dyx < 9; ++dyx) {
        const int dy = dyx / 3, dx = dyx % 3;
        const int yy = y + dy - 1;
        const bool rowok = (yy >= 0 && yy < 64);
        __builtin_prefetch(wt + ((size_t)dyx * 512 + m0) * 512, 0, 1);
        for (int c0 = 0; c0 < 512; c0 += 32) {
            // stage A: 64x32 weights, contiguous rows -> one 16B async copy per lane
            {
                int m  = tid >> 2;
                int cg = (tid & 3) * 8;
                stage16B(&As[m * STRH + cg],
                         wt + (((size_t)dyx * 512 + (m0 + m)) * 512) + c0 + cg);
            }
            // stage B: 32 channels x 64 shifted pixels, transposed into [n][c]
            {
                int cl = tid >> 3;
                int nb = (tid & 7) * 8;
                const _Float16* xrow =
                    xf + (((size_t)(b * 512 + c0 + cl) * 64 + (rowok ? yy : 0)) * 64);
#pragma unroll
                for (int j = 0; j < 8; ++j) {
                    int n = nb + j;
                    int xx = n + dx - 1;
                    _Float16 v = (_Float16)0.f;
                    if (rowok && xx >= 0 && xx < 64) v = xrow[xx];
                    Bs[n * STRH + cl] = v;
                }
            }
            stage_wait();
            __syncthreads();
            FragU fa, fb0, fb1;
            const int mrow  = mt * 16 + l16;
            const int nrow0 = (ng * 2 + 0) * 16 + l16;
            const int nrow1 = nrow0 + 16;
#pragma unroll
            for (int v = 0; v < 8; ++v) {
                fa.u[v]  = *reinterpret_cast<const unsigned*>(&As[mrow  * STRH + kmapA(v, half)]);
                fb0.u[v] = *reinterpret_cast<const unsigned*>(&Bs[nrow0 * STRH + kmapB(v, half)]);
                fb1.u[v] = *reinterpret_cast<const unsigned*>(&Bs[nrow1 * STRH + kmapB(v, half)]);
            }
            acc0 = __builtin_amdgcn_wmma_f32_16x16x32_f16(false, fa.v, false, fb0.v,
                                                          (short)0, acc0, false, false);
            acc1 = __builtin_amdgcn_wmma_f32_16x16x32_f16(false, fa.v, false, fb1.v,
                                                          (short)0, acc1, false, false);
            __syncthreads();
        }
    }
    // epilogue: bias + ReLU + store f16
#pragma unroll
    for (int r = 0; r < 8; ++r) {
        int mg = m0 + mt * 16 + r + half * 8;
        float bias = b1[mg];
        float v0 = acc0[r] + bias; v0 = v0 < 0.f ? 0.f : v0;
        float v1 = acc1[r] + bias; v1 = v1 < 0.f ? 0.f : v1;
        int n0 = (ng * 2 + 0) * 16 + l16;
        int n1 = (ng * 2 + 1) * 16 + l16;
        size_t base = ((size_t)(b * 512 + mg) * 64 + y) * 64;
        cf[base + n0] = (_Float16)v0;
        cf[base + n1] = (_Float16)v1;
    }
}

// ---------------- fused 1x1 convs (scores+locs) as WMMA GEMM ----------------
// grid: (64 pixel-tiles, 1, 8 images); M=64 (54 used), K=512
__global__ __launch_bounds__(256) void conv1x1_wmma_kernel(
    const _Float16* __restrict__ cf, const _Float16* __restrict__ wc,
    const float* __restrict__ bs, const float* __restrict__ bl,
    float* __restrict__ out_scores, float* __restrict__ out_locs) {
    __shared__ __align__(16) _Float16 As[64 * STRH];
    __shared__ __align__(16) _Float16 Bs[64 * STRH];
    const int tid  = threadIdx.x;
    const int pix0 = blockIdx.x * 64;
    const int b    = blockIdx.z;
    const int lane = tid & 31;
    const int wv   = tid >> 5;
    const int half = lane >> 4;
    const int l16  = lane & 15;
    const int mt   = wv & 3;
    const int ng   = wv >> 2;

    v8f acc0 = {}; v8f acc1 = {};

    for (int c0 = 0; c0 < 512; c0 += 32) {
        {
            int m  = tid >> 2;
            int cg = (tid & 3) * 8;
            stage16B(&As[m * STRH + cg], wc + (size_t)m * 512 + c0 + cg);
        }
        {
            int cl = tid >> 3;
            int nb = (tid & 7) * 8;
            const _Float16* crow = cf + ((size_t)(b * 512 + c0 + cl) * 4096) + pix0;
#pragma unroll
            for (int j = 0; j < 8; ++j) {
                int n = nb + j;
                Bs[n * STRH + cl] = crow[n];
            }
        }
        stage_wait();
        __syncthreads();
        FragU fa, fb0, fb1;
        const int mrow  = mt * 16 + l16;
        const int nrow0 = (ng * 2 + 0) * 16 + l16;
        const int nrow1 = nrow0 + 16;
#pragma unroll
        for (int v = 0; v < 8; ++v) {
            fa.u[v]  = *reinterpret_cast<const unsigned*>(&As[mrow  * STRH + kmapA(v, half)]);
            fb0.u[v] = *reinterpret_cast<const unsigned*>(&Bs[nrow0 * STRH + kmapB(v, half)]);
            fb1.u[v] = *reinterpret_cast<const unsigned*>(&Bs[nrow1 * STRH + kmapB(v, half)]);
        }
        acc0 = __builtin_amdgcn_wmma_f32_16x16x32_f16(false, fa.v, false, fb0.v,
                                                      (short)0, acc0, false, false);
        acc1 = __builtin_amdgcn_wmma_f32_16x16x32_f16(false, fa.v, false, fb1.v,
                                                      (short)0, acc1, false, false);
        __syncthreads();
    }
    // epilogue: scatter into transposed score/loc layouts
#pragma unroll
    for (int r = 0; r < 8; ++r) {
        int ch = mt * 16 + r + half * 8;
        if (ch >= 54) continue;
        float vals[2] = {acc0[r], acc1[r]};
        int pixs[2] = {pix0 + (ng * 2 + 0) * 16 + l16, pix0 + (ng * 2 + 1) * 16 + l16};
#pragma unroll
        for (int t = 0; t < 2; ++t) {
            int pix = pixs[t];
            if (ch < 18) {
                float v = vals[t] + bs[ch];
                out_scores[((size_t)b * 36864 + (size_t)pix * 9 + (ch >> 1)) * 2 + (ch & 1)] = v;
            } else {
                int cl = ch - 18;
                float v = vals[t] + bl[cl];
                out_locs[((size_t)b * 36864 + (size_t)pix * 9 + (cl >> 2)) * 4 + (cl & 3)] = v;
            }
        }
    }
}

// ---------------- softmax + box decode + clip + size filter ----------------
__global__ void decode_kernel(const float* __restrict__ out_scores, const float* __restrict__ out_locs,
                              const float* __restrict__ anch,
                              const int* __restrict__ ph, const int* __restrict__ pw,
                              float* __restrict__ boxes, float* __restrict__ s, int* __restrict__ sup) {
    int idx = blockIdx.x * blockDim.x + threadIdx.x;
    if (idx >= 8 * 36864) return;
    int i = idx % 36864;
    float s0 = out_scores[(size_t)idx * 2 + 0];
    float s1 = out_scores[(size_t)idx * 2 + 1];
    float m = fmaxf(s0, s1);
    float e0 = expf(s0 - m), e1 = expf(s1 - m);
    float fg = e1 / (e0 + e1);
    const float* l = out_locs + (size_t)idx * 4;
    float a0 = anch[(size_t)i * 4 + 0], a1 = anch[(size_t)i * 4 + 1];
    float a2 = anch[(size_t)i * 4 + 2], a3 = anch[(size_t)i * 4 + 3];
    float ha = a2 - a0, wa = a3 - a1;
    float cya = a0 + 0.5f * ha, cxa = a1 + 0.5f * wa;
    float cy = l[0] * ha + cya, cx = l[1] * wa + cxa;
    float hb = expf(l[2]) * ha, wb = expf(l[3]) * wa;
    float ih = (float)ph[0], iw = (float)pw[0];
    float y1 = fminf(fmaxf(cy - 0.5f * hb, 0.f), ih);
    float x1 = fminf(fmaxf(cx - 0.5f * wb, 0.f), iw);
    float y2 = fminf(fmaxf(cy + 0.5f * hb, 0.f), ih);
    float x2 = fminf(fmaxf(cx + 0.5f * wb, 0.f), iw);
    boxes[(size_t)idx * 4 + 0] = y1;
    boxes[(size_t)idx * 4 + 1] = x1;
    boxes[(size_t)idx * 4 + 2] = y2;
    boxes[(size_t)idx * 4 + 3] = x2;
    bool valid = (y2 - y1 >= 16.f) && (x2 - x1 >= 16.f);
    s[idx] = valid ? fg : -INFINITY;
    sup[idx] = 0;
}

// ---------------- greedy NMS: one block per image ----------------
__global__ __launch_bounds__(256) void nms_kernel(const float* __restrict__ boxes,
                                                  const float* __restrict__ s,
                                                  int* __restrict__ sup,
                                                  float* __restrict__ rois,
                                                  float* __restrict__ inds) {
    const int b = blockIdx.x;
    const int tid = threadIdx.x;
    const int N = 36864, PT = 144;   // 256*144 == 36864
    const float* bx = boxes + (size_t)b * N * 4;
    const float* sc = s + (size_t)b * N;
    int* sp = sup + (size_t)b * N;
    __shared__ float rv[256];
    __shared__ int   ri[256];
    __shared__ float jb[6];   // y1 x1 y2 x2 area ok
    const int base = tid * PT;
    for (int it = 0; it < 2000; ++it) {
        float bestv = -INFINITY; int besti = -1;
        for (int k = 0; k < PT; ++k) {
            int i = base + k;
            if (!sp[i]) {
                float v = sc[i];
                if (v > bestv) { bestv = v; besti = i; }   // strict > keeps first max
            }
        }
        rv[tid] = bestv; ri[tid] = besti;
        __syncthreads();
        for (int off = 128; off > 0; off >>= 1) {
            if (tid < off) {
                float v2 = rv[tid + off]; int i2 = ri[tid + off];
                if (v2 > rv[tid]) { rv[tid] = v2; ri[tid] = i2; }
            }
            __syncthreads();
        }
        if (tid == 0) {
            int j = ri[0];
            bool ok = (j >= 0) && (rv[0] != -INFINITY);
            size_t ro = ((size_t)b * 2000 + it) * 4;
            if (ok) {
                float y1 = bx[(size_t)j * 4 + 0], x1 = bx[(size_t)j * 4 + 1];
                float y2 = bx[(size_t)j * 4 + 2], x2 = bx[(size_t)j * 4 + 3];
                jb[0] = y1; jb[1] = x1; jb[2] = y2; jb[3] = x2;
                jb[4] = (x2 - x1 + 1.f) * (y2 - y1 + 1.f);
                jb[5] = 1.f;
                rois[ro + 0] = y1; rois[ro + 1] = x1; rois[ro + 2] = y2; rois[ro + 3] = x2;
                inds[(size_t)b * 2000 + it] = (float)b;
            } else {
                jb[5] = 0.f;
                rois[ro + 0] = 0.f; rois[ro + 1] = 0.f; rois[ro + 2] = 0.f; rois[ro + 3] = 0.f;
                inds[(size_t)b * 2000 + it] = -1.f;
            }
        }
        __syncthreads();
        if (jb[5] > 0.f) {
            float y1j = jb[0], x1j = jb[1], y2j = jb[2], x2j = jb[3], aj = jb[4];
            for (int k = 0; k < PT; ++k) {
                int i = base + k;
                if (sp[i]) continue;
                float y1 = bx[(size_t)i * 4 + 0], x1 = bx[(size_t)i * 4 + 1];
                float y2 = bx[(size_t)i * 4 + 2], x2 = bx[(size_t)i * 4 + 3];
                float ymin = fmaxf(y1, y1j), xmin = fmaxf(x1, x1j);
                float ymax = fminf(y2, y2j), xmax = fminf(x2, x2j);
                float inter = fmaxf(xmax - xmin, 0.f) * fmaxf(ymax - ymin, 0.f);
                float area = (x2 - x1 + 1.f) * (y2 - y1 + 1.f);
                float iou = inter / (aj + area - inter);
                if (iou > 0.7f) sp[i] = 1;
            }
        }
        __syncthreads();
    }
}

// ---------------- launcher ----------------
extern "C" void kernel_launch(void* const* d_in, const int* in_sizes, int n_in,
                              void* d_out, int out_size, void* d_ws, size_t ws_size,
                              hipStream_t stream) {
    const float* x  = (const float*)d_in[0];
    const float* w1 = (const float*)d_in[1];
    const float* b1 = (const float*)d_in[2];
    const float* wsq = (const float*)d_in[3];
    const float* bsv = (const float*)d_in[4];
    const float* wlq = (const float*)d_in[5];
    const float* blv = (const float*)d_in[6];
    const int* ph = (const int*)d_in[7];
    const int* pw = (const int*)d_in[8];
    float* out = (float*)d_out;

    // output regions (float elements)
    float* o_rois    = out;                 // 8*2000*4    = 64000
    float* o_inds    = out + 64000;         // 8*2000      = 16000
    float* o_anchors = out + 80000;         // 36864*4     = 147456
    float* o_scores  = out + 227456;        // 8*36864*2   = 589824
    float* o_locs    = out + 817280;        // 8*36864*4   = 1179648

    // workspace layout (bytes)
    char* wsb = (char*)d_ws;
    _Float16* xf  = (_Float16*)(wsb);                       // 33,554,432 B
    _Float16* wt  = (_Float16*)(wsb + 33554432u);           //  4,718,592 B  [dyx][k][c]
    _Float16* cf  = (_Float16*)(wsb + 38273024u);           // 33,554,432 B
    _Float16* wc  = (_Float16*)(wsb + 71827456u);           //     65,536 B
    float*    boxes = (float*)(wsb + 71892992u);            //  4,718,592 B
    float*    sarr  = (float*)(wsb + 76611584u);            //  1,179,648 B
    int*      sup   = (int*)  (wsb + 77791232u);            //  1,179,648 B

    const int NX = 8 * 512 * 64 * 64;       // 16,777,216
    const int NW = 512 * 512 * 9;           //  2,359,296

    f32_to_f16_kernel<<<NX / 256, 256, 0, stream>>>(x, xf, NX);
    transform_w1_kernel<<<NW / 256, 256, 0, stream>>>(w1, wt);
    build_wc_kernel<<<128, 256, 0, stream>>>(wsq, wlq, wc);
    anchors_kernel<<<144, 256, 0, stream>>>(o_anchors);

    conv3x3_wmma_kernel<<<dim3(8, 64, 8), 256, 0, stream>>>(xf, wt, b1, cf);
    conv1x1_wmma_kernel<<<dim3(64, 1, 8), 256, 0, stream>>>(cf, wc, bsv, blv, o_scores, o_locs);

    decode_kernel<<<1152, 256, 0, stream>>>(o_scores, o_locs, o_anchors, ph, pw, boxes, sarr, sup);
    nms_kernel<<<8, 256, 0, stream>>>(boxes, sarr, sup, o_rois, o_inds);
}
